// SpanPredictor_86431921864706
// MI455X (gfx1250) — compile-verified, matrix operand order checked by reading, not run on hardware
//
#include <hip/hip_runtime.h>

// ---------------------------------------------------------------------------
// SpanPredictor for MI455X (gfx1250, wave32, WMMA).
// Heavy math restructured into shared f16 WMMA GEMMs with f32 accumulation.
// B (weight) matrices are pre-packed into the CDNA5 WMMA B-fragment lane
// layout; GEMM waves compute 32x64 tiles with double-buffered fragments:
// 8 v_wmma per K-step fed by 12 b128 loads (1.5 loads/wmma).
// ---------------------------------------------------------------------------

#define N_WORDS   8176
#define N_WORDS_P 8192        // padded to a multiple of 32 rows for the GEMM
#define N_HEADS   512
#define EMB       768
#define DIST      64
#define NUM_SENTS 146
#define MAX_LEN   64
#define HID2      256
#define HID3      64
#define ROWS      (N_HEADS * MAX_LEN)   // 32768

typedef __attribute__((ext_vector_type(16))) _Float16 v16h;
typedef __attribute__((ext_vector_type(8)))  _Float16 v8h;
typedef __attribute__((ext_vector_type(8)))  float    v8f;

// ---------------------------------------------------------------------------
// Fill output with -inf (grid-stride, float4 stores)
// ---------------------------------------------------------------------------
__global__ __launch_bounds__(256)
void sp_fill_neginf(float4* __restrict__ out, size_t n4) {
  float4 v;
  v.x = v.y = v.z = v.w = -INFINITY;
  for (size_t i = (size_t)blockIdx.x * blockDim.x + threadIdx.x; i < n4;
       i += (size_t)gridDim.x * blockDim.x)
    out[i] = v;
}

// ---------------------------------------------------------------------------
// Sentence metadata: sent_start (first word index) and lengths. One block.
// ---------------------------------------------------------------------------
__global__ __launch_bounds__(256)
void sp_prep(const int* __restrict__ sent_id,
             int* __restrict__ sent_start, int* __restrict__ lengths) {
  for (int i = threadIdx.x; i < N_WORDS; i += blockDim.x) {
    if (i == 0)                          sent_start[sent_id[0]] = 0;
    else if (sent_id[i] != sent_id[i-1]) sent_start[sent_id[i]] = i;
  }
  __syncthreads();
  for (int s = threadIdx.x; s < NUM_SENTS; s += blockDim.x) {
    int nxt = (s == NUM_SENTS - 1) ? N_WORDS : sent_start[s + 1];
    lengths[s] = nxt - sent_start[s];
  }
}

// ---------------------------------------------------------------------------
// f32 -> f16 convert (grid-stride), row-major kept (A matrices)
// ---------------------------------------------------------------------------
__global__ __launch_bounds__(256)
void sp_cvt_f16(const float* __restrict__ src, _Float16* __restrict__ dst, size_t n) {
  for (size_t i = (size_t)blockIdx.x * blockDim.x + threadIdx.x; i < n;
       i += (size_t)gridDim.x * blockDim.x)
    dst[i] = (_Float16)src[i];
}

// ---------------------------------------------------------------------------
// Gather word rows for heads, convert to f16. One block per head.
// ---------------------------------------------------------------------------
__global__ __launch_bounds__(256)
void sp_gather_rows_f16(const float* __restrict__ src, const int* __restrict__ idx,
                        _Float16* __restrict__ dst, int ld) {
  int r = blockIdx.x;
  size_t srow = (size_t)idx[r] * ld;
  size_t drow = (size_t)r * ld;
  for (int c = threadIdx.x; c < ld; c += blockDim.x)
    dst[drow + c] = (_Float16)src[srow + c];
}

// ---------------------------------------------------------------------------
// Pack a row-major f32 [K x N] matrix into WMMA B-fragment layout (f16):
//   dst[((kt*nTiles + nt)*32 + lane)*16 + e]
// CDNA5 ISA 7.12.2, 16-bit 32x16 B fragment: lane = N column, lane-half
// selects K-octet: kb=(lane>>4)*8; e<8 -> k=kt*32+kb+e ; e>=8 -> +8 more.
// ---------------------------------------------------------------------------
__global__ __launch_bounds__(256)
void sp_pack_b(const float* __restrict__ src, _Float16* __restrict__ dst,
               int K, int N) {
  const int nTiles = N >> 4;
  const size_t total = (size_t)K * N;
  for (size_t i = (size_t)blockIdx.x * blockDim.x + threadIdx.x; i < total;
       i += (size_t)gridDim.x * blockDim.x) {
    const int e    = (int)(i & 15);
    const int lane = (int)((i >> 4) & 31);
    const size_t t = i >> 9;              // kt*nTiles + nt
    const int nt = (int)(t % (size_t)nTiles);
    const int kt = (int)(t / (size_t)nTiles);
    const int kb = (lane >> 4) * 8;
    const int k  = kt * 32 + kb + e + ((e >= 8) ? 8 : 0);
    const int n  = nt * 16 + (lane & 15);
    dst[i] = (_Float16)src[(size_t)k * N + n];
  }
}

// ---------------------------------------------------------------------------
// A-fragment load: 16x32 f16 tile row slice for one lane (two 16B loads).
// ---------------------------------------------------------------------------
__device__ __forceinline__ v16h sp_ldA(const _Float16* __restrict__ p) {
  const v8h lo = *(const v8h*)p;
  const v8h hi = *(const v8h*)(p + 16);
  return __builtin_shufflevector(lo, hi,
      0, 1, 2, 3, 4, 5, 6, 7, 8, 9, 10, 11, 12, 13, 14, 15);
}

#define SP_WMMA(a_, b_, c_) \
  __builtin_amdgcn_wmma_f32_16x16x32_f16(false, (a_), false, (b_), (short)0, \
                                         (c_), false, false)

// ---------------------------------------------------------------------------
// Generic WMMA GEMM:  C[M,N] = op(A[M,K] @ B[K,N] + bias)
//   A: f16 row-major (M % 32 == 0).  Bp: f16 pre-packed fragments.
//   f32 accumulate; optional f32 and/or f16 outputs, optional bias + ReLU.
// Wave tile: 32(M) x 64(N); block (8 waves) tile: 256(M) x 64(N).
// Double-buffered: next K-step fragments prefetched before current WMMAs.
// ---------------------------------------------------------------------------
__global__ __launch_bounds__(256)
void sp_gemm_f16(const _Float16* __restrict__ A, const _Float16* __restrict__ Bp,
                 float* __restrict__ Cf, _Float16* __restrict__ Ch,
                 const float* __restrict__ bias,
                 int M, int N, int K, int relu) {
  const int lane = threadIdx.x & 31;
  const int wave = threadIdx.x >> 5;
  const int m0 = (blockIdx.x * 8 + wave) * 32;   // 32 rows per wave
  if (m0 >= M) return;
  const int nBase  = blockIdx.y * 64;
  const int nt0    = nBase >> 4;
  const int nTiles = N >> 4;

  const int ml = lane & 15;        // A: row within tile / B,C: column within tile
  const int kh = lane >> 4;        // lane-half selects K-octet (and C row group)

  const v16h* __restrict__ bp = (const v16h*)Bp + (size_t)nt0 * 32 + lane;
  const size_t bStep = (size_t)nTiles * 32;      // v16h per K-step

  const _Float16* __restrict__ apA = A + (size_t)(m0 + ml) * K + kh * 8;
  const _Float16* __restrict__ apB = apA + (size_t)16 * K;

  v8f acc[8] = {};

  // prologue: fragments for k0 = 0
  v16h a0 = sp_ldA(apA);
  v16h a1 = sp_ldA(apB);
  v16h b0 = bp[0], b1 = bp[32], b2 = bp[64], b3 = bp[96];

  for (int k0 = 32; k0 < K; k0 += 32) {
    // prefetch next K-step
    const v16h na0 = sp_ldA(apA + k0);
    const v16h na1 = sp_ldA(apB + k0);
    const v16h* nb = bp + (size_t)(k0 >> 5) * bStep;
    const v16h nb0 = nb[0], nb1 = nb[32], nb2 = nb[64], nb3 = nb[96];

    acc[0] = SP_WMMA(a0, b0, acc[0]);
    acc[1] = SP_WMMA(a0, b1, acc[1]);
    acc[2] = SP_WMMA(a0, b2, acc[2]);
    acc[3] = SP_WMMA(a0, b3, acc[3]);
    acc[4] = SP_WMMA(a1, b0, acc[4]);
    acc[5] = SP_WMMA(a1, b1, acc[5]);
    acc[6] = SP_WMMA(a1, b2, acc[6]);
    acc[7] = SP_WMMA(a1, b3, acc[7]);

    a0 = na0; a1 = na1; b0 = nb0; b1 = nb1; b2 = nb2; b3 = nb3;
  }
  // final K-step
  acc[0] = SP_WMMA(a0, b0, acc[0]);
  acc[1] = SP_WMMA(a0, b1, acc[1]);
  acc[2] = SP_WMMA(a0, b2, acc[2]);
  acc[3] = SP_WMMA(a0, b3, acc[3]);
  acc[4] = SP_WMMA(a1, b0, acc[4]);
  acc[5] = SP_WMMA(a1, b1, acc[5]);
  acc[6] = SP_WMMA(a1, b2, acc[6]);
  acc[7] = SP_WMMA(a1, b3, acc[7]);

  // ---- epilogue: C/D f32 layout: VGPR v <-> row base + v + 8*kh, col n0+ml
#pragma unroll
  for (int half = 0; half < 2; ++half) {
    const int rbase = m0 + half * 16 + 8 * kh;
#pragma unroll
    for (int nt = 0; nt < 4; ++nt) {
      const v8f c = acc[half * 4 + nt];
      const int n = nBase + nt * 16 + ml;
      const float bv = bias ? bias[n] : 0.0f;
#pragma unroll
      for (int v = 0; v < 8; ++v) {
        float x = c[v] + bv;
        if (relu) x = fmaxf(x, 0.0f);
        if (Cf) Cf[(size_t)(rbase + v) * N + n] = x;
        if (Ch) Ch[(size_t)(rbase + v) * N + n] = (_Float16)x;
      }
    }
  }
}

// ---------------------------------------------------------------------------
// Layer-1 assembly: h1[h,l,:] = relu(HeadProj[h] + ColProj[col] + EmbProj[eid]
//                                    + b1)  (padded slot: relu(b1)), f16 out.
// ---------------------------------------------------------------------------
__global__ __launch_bounds__(256)
void sp_assemble_h1(const int* __restrict__ heads_ids, const int* __restrict__ sent_id,
                    const int* __restrict__ sent_start, const int* __restrict__ lengths,
                    const float* __restrict__ HeadProj, const float* __restrict__ ColProj,
                    const float* __restrict__ EmbProj, const float* __restrict__ b1,
                    _Float16* __restrict__ h1) {
  const int h = blockIdx.x;
  const int hid   = heads_ids[h];
  const int s     = sent_id[hid];
  const int start = sent_start[s];
  const int len   = lengths[s];
  const size_t hbase = (size_t)h * EMB;
  for (int idx = threadIdx.x; idx < MAX_LEN * EMB; idx += blockDim.x) {
    const int l = idx / EMB;
    const int f = idx - l * EMB;
    float v;
    if (l < len) {
      const int col = start + l;
      int e = (hid - col) + 63;
      if (e < 0 || e > 126) e = 127;
      v = HeadProj[hbase + f] + ColProj[(size_t)col * EMB + f] +
          EmbProj[(size_t)e * EMB + f] + b1[f];
    } else {
      v = b1[f];
    }
    h1[((size_t)h * MAX_LEN + l) * EMB + f] = (_Float16)fmaxf(v, 0.0f);
  }
}

// ---------------------------------------------------------------------------
// Per-head: conv1d(64->4,k3,SAME) -> conv1d(4->2,k3,SAME) over L=64 axis,
// then masked scatter into [H, n_words, 2] with start/end validity masks.
// ---------------------------------------------------------------------------
__global__ __launch_bounds__(256)
void sp_conv_scatter(const float* __restrict__ h3,
                     const float* __restrict__ cw1, const float* __restrict__ cb1,
                     const float* __restrict__ cw2, const float* __restrict__ cb2,
                     const int* __restrict__ heads_ids, const int* __restrict__ sent_id,
                     const int* __restrict__ sent_start, const int* __restrict__ lengths,
                     float* __restrict__ out) {
  __shared__ float sh3[MAX_LEN + 2][HID3];   // zero-padded rows l=-1..64
  __shared__ float st1[MAX_LEN + 2][4];
  const int h = blockIdx.x;

  for (int idx = threadIdx.x; idx < MAX_LEN * HID3; idx += blockDim.x) {
    const int l = idx >> 6, c = idx & 63;
    sh3[l + 1][c] = h3[((size_t)h * MAX_LEN + l) * HID3 + c];
  }
  if (threadIdx.x < HID3) {
    sh3[0][threadIdx.x] = 0.0f;
    sh3[MAX_LEN + 1][threadIdx.x] = 0.0f;
  }
  if (threadIdx.x < 4) {
    st1[0][threadIdx.x] = 0.0f;
    st1[MAX_LEN + 1][threadIdx.x] = 0.0f;
  }
  __syncthreads();

  {   // conv1: 64 l x 4 oc == 256 threads
    const int l = threadIdx.x >> 2, oc = threadIdx.x & 3;
    float a = cb1[oc];
#pragma unroll
    for (int k = 0; k < 3; ++k)
      for (int ic = 0; ic < HID3; ++ic)
        a += sh3[l + k][ic] * cw1[((size_t)k * HID3 + ic) * 4 + oc];
    st1[l + 1][oc] = a;
  }
  __syncthreads();

  if (threadIdx.x < 2 * MAX_LEN) {   // conv2 + scatter: 64 l x 2 oc
    const int l = threadIdx.x >> 1, oc = threadIdx.x & 1;
    float a = cb2[oc];
#pragma unroll
    for (int k = 0; k < 3; ++k)
      for (int ic = 0; ic < 4; ++ic)
        a += st1[l + k][ic] * cw2[((size_t)k * 4 + ic) * 2 + oc];

    const int hid   = heads_ids[h];
    const int s     = sent_id[hid];
    const int start = sent_start[s];
    const int len   = lengths[s];
    if (l < len) {
      const int col = start + l;
      const int rel = hid - col;
      const bool valid = (oc == 0) ? (rel >= 0) : (rel <= 0);
      out[((size_t)h * N_WORDS + col) * 2 + oc] = valid ? a : -INFINITY;
    }
  }
}

// ---------------------------------------------------------------------------
// Host-side orchestration
// ---------------------------------------------------------------------------
extern "C" void kernel_launch(void* const* d_in, const int* in_sizes, int n_in,
                              void* d_out, int out_size, void* d_ws, size_t ws_size,
                              hipStream_t stream) {
  (void)in_sizes; (void)n_in; (void)ws_size;
  const float* words     = (const float*)d_in[0];
  const int*   heads_ids = (const int*)  d_in[1];
  const int*   sent_id   = (const int*)  d_in[2];
  const float* emb_table = (const float*)d_in[3];
  const float* W1        = (const float*)d_in[4];
  const float* b1        = (const float*)d_in[5];
  const float* W2        = (const float*)d_in[6];
  const float* b2        = (const float*)d_in[7];
  const float* W3        = (const float*)d_in[8];
  const float* b3        = (const float*)d_in[9];
  const float* cw1       = (const float*)d_in[10];
  const float* cb1       = (const float*)d_in[11];
  const float* cw2       = (const float*)d_in[12];
  const float* cb2       = (const float*)d_in[13];
  float* out = (float*)d_out;

  // ---- scratch carve-out (256B aligned)
  char* wptr = (char*)d_ws;
  auto carve = [&](size_t bytes) -> void* {
    void* p = (void*)wptr;
    wptr += (bytes + 255) & ~(size_t)255;
    return p;
  };
  int*       ss_start = (int*)carve(NUM_SENTS * sizeof(int));
  int*       ss_len   = (int*)carve(NUM_SENTS * sizeof(int));
  _Float16*  words_h  = (_Float16*)carve((size_t)N_WORDS_P * EMB * 2);  // A (padded)
  _Float16*  emb_h    = (_Float16*)carve((size_t)128 * DIST * 2);       // A
  _Float16*  headw_h  = (_Float16*)carve((size_t)N_HEADS * EMB * 2);    // A
  _Float16*  W1a_p    = (_Float16*)carve((size_t)EMB * EMB * 2);        // B packed
  _Float16*  W1b_p    = (_Float16*)carve((size_t)EMB * EMB * 2);
  _Float16*  W1c_p    = (_Float16*)carve((size_t)DIST * EMB * 2);
  _Float16*  W2_p     = (_Float16*)carve((size_t)EMB * HID2 * 2);
  _Float16*  W3_p     = (_Float16*)carve((size_t)HID2 * HID3 * 2);
  float*     ColProj  = (float*)carve((size_t)N_WORDS_P * EMB * 4);     // padded M
  float*     HeadProj = (float*)carve((size_t)N_HEADS * EMB * 4);
  float*     EmbProj  = (float*)carve((size_t)128 * EMB * 4);
  _Float16*  h1_h     = (_Float16*)carve((size_t)ROWS * EMB * 2);
  _Float16*  h2_h     = (_Float16*)carve((size_t)ROWS * HID2 * 2);
  float*     h3_f     = (float*)carve((size_t)ROWS * HID3 * 4);

  // 1) -inf fill of output
  {
    size_t n4 = (size_t)out_size / 4;
    sp_fill_neginf<<<4096, 256, 0, stream>>>((float4*)out, n4);
  }
  // 2) sentence metadata
  sp_prep<<<1, 256, 0, stream>>>(sent_id, ss_start, ss_len);

  // 3) A-side f16 conversions + head-row gather
  sp_cvt_f16<<<2048, 256, 0, stream>>>(words, words_h, (size_t)N_WORDS * EMB);
  sp_cvt_f16<<<8,    256, 0, stream>>>(emb_table, emb_h, (size_t)128 * DIST);
  sp_gather_rows_f16<<<N_HEADS, 256, 0, stream>>>(words, heads_ids, headw_h, EMB);

  // 3b) B-side weight packing into WMMA fragment layout.
  //     W1 rows: [0:768]=head part, [768:1536]=col part, [1536:1600]=emb part.
  sp_pack_b<<<1024, 256, 0, stream>>>(W1,                         W1a_p, EMB,  EMB);
  sp_pack_b<<<1024, 256, 0, stream>>>(W1 + (size_t)EMB * EMB,     W1b_p, EMB,  EMB);
  sp_pack_b<<<96,   256, 0, stream>>>(W1 + (size_t)2 * EMB * EMB, W1c_p, DIST, EMB);
  sp_pack_b<<<384,  256, 0, stream>>>(W2,                         W2_p,  EMB,  HID2);
  sp_pack_b<<<32,   256, 0, stream>>>(W3,                         W3_p,  HID2, HID3);

  // 4) shared layer-1 projections (WMMA)
  // ColProj = words @ W1b : M=8192 padded (gx=32), N=768 (gy=12), K=768
  sp_gemm_f16<<<dim3(32, 12), 256, 0, stream>>>(
      words_h, W1b_p, ColProj, nullptr, nullptr, N_WORDS_P, EMB, EMB, 0);
  // HeadProj = words[heads] @ W1a : M=512 (gx=2), N=768, K=768
  sp_gemm_f16<<<dim3(2, 12), 256, 0, stream>>>(
      headw_h, W1a_p, HeadProj, nullptr, nullptr, N_HEADS, EMB, EMB, 0);
  // EmbProj = emb_table @ W1c : M=128 (gx=1), N=768, K=64
  sp_gemm_f16<<<dim3(1, 12), 256, 0, stream>>>(
      emb_h, W1c_p, EmbProj, nullptr, nullptr, 128, EMB, DIST, 0);

  // 5) layer-1 assembly (gather + bias + ReLU -> f16)
  sp_assemble_h1<<<N_HEADS, 256, 0, stream>>>(
      heads_ids, sent_id, ss_start, ss_len, HeadProj, ColProj, EmbProj, b1, h1_h);

  // 6) layer 2: h2 = relu(h1 @ W2 + b2)  M=32768 (gx=128), N=256 (gy=4), K=768
  sp_gemm_f16<<<dim3(128, 4), 256, 0, stream>>>(
      h1_h, W2_p, nullptr, h2_h, b2, ROWS, HID2, EMB, 1);

  // 7) layer 3: h3 = h2 @ W3 + b3        M=32768 (gx=128), N=64 (gy=1), K=256
  sp_gemm_f16<<<dim3(128, 1), 256, 0, stream>>>(
      h2_h, W3_p, h3_f, nullptr, b3, ROWS, HID3, HID2, 0);

  // 8) conv1d x2 + masked scatter into the -inf canvas
  sp_conv_scatter<<<N_HEADS, 256, 0, stream>>>(
      h3_f, cw1, cb1, cw2, cb2, heads_ids, sent_id, ss_start, ss_len, out);
}
